// Node_GCN_4252017623285
// MI455X (gfx1250) — matrix operand order repaired and checked by reading, not compiled, not run on hardware
//
#include <hip/hip_runtime.h>
#include <hip/hip_fp16.h>

// Node_GCN forward for MI455X (gfx1250, wave32).
//   r0 = relu(X @ We)                               [WMMA f16->f32, 80-row tiles]
//   hop: s = A r; t = A s; r' = relu([s-r, t-s-r])  [edge-parallel float4 atomic SpMM]
//   out = softmax(concat(r0,r1,r2) @ Wc)            [WMMA f16->f32 + fused softmax]
// Weights are pre-fragmented once into f16 WMMA B-fragment order in global ws,
// so GEMM blocks only stage the A tile through LDS.

#define GCN_N      50000
#define GCN_FEAT   256
#define GCN_E      800000

typedef __attribute__((ext_vector_type(16))) _Float16 v16h;
typedef __attribute__((ext_vector_type(4)))  _Float16 v4h;
typedef __attribute__((ext_vector_type(8)))  float    v8f;

// ---------------------------------------------------------------- utilities
__global__ __launch_bounds__(256) void zero_f32v4(float4* __restrict__ p, int n4) {
  int i = blockIdx.x * blockDim.x + threadIdx.x;
  if (i < n4) p[i] = float4{0.f, 0.f, 0.f, 0.f};
}

// out[rows[e], f..f+3] += vals[e] * in[cols[e], f..f+3]
template <int D>
__global__ __launch_bounds__(256)
void spmm_atomic(const int* __restrict__ rows, const int* __restrict__ cols,
                 const float* __restrict__ vals, const float* __restrict__ in,
                 float* __restrict__ out) {
  constexpr int DQ = D / 4;
  int idx = blockIdx.x * 256 + threadIdx.x;
  if (idx >= GCN_E * DQ) return;
  int e = idx / DQ;
  int f = (idx - e * DQ) << 2;
  float v = vals[e];
  float4 g = *(const float4*)&in[(size_t)cols[e] * D + f];
  float* o = &out[(size_t)rows[e] * D + f];
  unsafeAtomicAdd(o + 0, v * g.x);   // global_atomic_add_f32
  unsafeAtomicAdd(o + 1, v * g.y);
  unsafeAtomicAdd(o + 2, v * g.z);
  unsafeAtomicAdd(o + 3, v * g.w);
}

// rn[:, :D] = relu(s - r) ; rn[:, D:2D] = relu(t - s - r)
template <int D>
__global__ __launch_bounds__(256)
void hop_combine(const float* __restrict__ r, const float* __restrict__ s,
                 const float* __restrict__ t, float* __restrict__ rn) {
  constexpr int DQ = D / 4;
  int idx = blockIdx.x * 256 + threadIdx.x;
  if (idx >= GCN_N * DQ) return;
  int i = idx / DQ;
  int f = (idx - i * DQ) << 2;
  size_t o = (size_t)i * D + f;
  float4 rv = *(const float4*)&r[o];
  float4 sv = *(const float4*)&s[o];
  float4 tv = *(const float4*)&t[o];
  float4 loq, hiq;
  loq.x = fmaxf(sv.x - rv.x, 0.f); hiq.x = fmaxf(tv.x - sv.x - rv.x, 0.f);
  loq.y = fmaxf(sv.y - rv.y, 0.f); hiq.y = fmaxf(tv.y - sv.y - rv.y, 0.f);
  loq.z = fmaxf(sv.z - rv.z, 0.f); hiq.z = fmaxf(tv.z - sv.z - rv.z, 0.f);
  loq.w = fmaxf(sv.w - rv.w, 0.f); hiq.w = fmaxf(tv.w - sv.w - rv.w, 0.f);
  *(float4*)&rn[(size_t)i * (2 * D) + f]     = loq;
  *(float4*)&rn[(size_t)i * (2 * D) + D + f] = hiq;
}

// --------------------------------------------------- fragment-order layouts
// A 16x32 f16 frag (ISA 7.12.2): reader lane L=(lo<<4)|m, element e:
//   v=e>>1, k = (v<4?0:16) + lo*8 + 2*(v&3) + (e&1)
// Inverse: lo=(k>>3)&1, e_base(k multiple of 4) = (((k>>4)<<2)|((k&7)>>1))<<1,
// and 4 consecutive k map to the same lane, consecutive e.
// B 32x16 frag: reader lane L=(lo<<4)|n, element e -> W[k0+lo*8+e+(e>>3)*8][wave*16+n] (same k order as A).

// Pre-fragment a [K x 64] f32 weight into f16 B-fragment order:
// Wf[((kt*4 + wave)*32 + lane)*16 + e]
__global__ __launch_bounds__(256)
void prep_wfrag(const float* __restrict__ W, _Float16* __restrict__ Wf, int K) {
  int idx = blockIdx.x * 256 + threadIdx.x;
  if (idx >= K * 64) return;
  int e    = idx & 15;
  int lane = (idx >> 4) & 31;
  int wv   = (idx >> 9) & 3;
  int kt   = idx >> 11;
  int k = kt * 32 + ((lane >> 4) << 3) + e + ((e >> 3) << 3);
  int c = (wv << 4) | (lane & 15);
  Wf[idx] = (_Float16)W[(size_t)k * 64 + c];
}

// A-tile loader: 80x32 f32 tile -> f16 fragment order in LDS.
// Uniform 5-iteration loop; float4 read -> 2x cvt_pk -> ds_store_b64.
__device__ inline void load_Atile(_Float16* sAf, const float* __restrict__ src,
                                  int row0, int ldshift, int kb, int tid) {
#pragma unroll
  for (int it = 0; it < 5; ++it) {
    int jj = tid + it * 128;
    int r = jj >> 3;               // tile row 0..79
    int c = (jj & 7) << 2;         // k offset in tile, multiple of 4
    float4 f = *(const float4*)&src[((size_t)(row0 + r) << ldshift) + kb + c];
    int st = r >> 4, m = r & 15;
    int lane_d = (((c >> 3) & 1) << 4) | m;
    int eb = (((c >> 4) << 2) | ((c & 7) >> 1)) << 1;   // multiple of 4
    v4h h;
    h[0] = (_Float16)f.x; h[1] = (_Float16)f.y;
    h[2] = (_Float16)f.z; h[3] = (_Float16)f.w;
    *(v4h*)&sAf[(((unsigned)st << 5) | lane_d) * 16 + eb] = h;
  }
}

// Per-k-step WMMA body: 1 global B fragment reused by 5 M-subtiles.
__device__ inline void wmma_5x(const _Float16* sAf, const v16h b,
                               int lane, v8f acc[5]) {
#pragma unroll
  for (int st = 0; st < 5; ++st) {
    v16h a = *(const v16h*)&sAf[(((unsigned)st << 5) | lane) * 16];
    acc[st] = __builtin_amdgcn_wmma_f32_16x16x32_f16(
        false, a, false, b, (short)0, acc[st], false, false);
  }
}

// ---------------------------- GEMM1: r0 = relu(X @ We), 80 rows per block
__global__ __launch_bounds__(128)
void gemm_embed_relu(const float* __restrict__ A, const _Float16* __restrict__ Wf,
                     float* __restrict__ C) {
  __shared__ __align__(32) _Float16 sAf[5 * 32 * 16];
  const int tid = threadIdx.x, wave = tid >> 5, lane = tid & 31;
  const int row0 = blockIdx.x * 80;            // 50000 = 80*625, no tail
  v8f acc[5] = {};
  for (int kt = 0; kt < GCN_FEAT / 32; ++kt) {
    load_Atile(sAf, A, row0, /*ldshift=*/8, kt * 32, tid);
    v16h b = *(const v16h*)&Wf[(((unsigned)(kt * 4 + wave)) * 32 + lane) * 16];
    __syncthreads();
    wmma_5x(sAf, b, lane, acc);
    __syncthreads();
  }
  const int lo = lane >> 4, mr = lane & 15;
#pragma unroll
  for (int st = 0; st < 5; ++st)
#pragma unroll
    for (int e = 0; e < 8; ++e) {
      int m = st * 16 + e + (lo ? 8 : 0);
      float v = acc[st][e];
      C[(size_t)(row0 + m) * 64 + (wave << 4) + mr] = v > 0.f ? v : 0.f;
    }
}

// ---------- GEMM2 + softmax: out = softmax(concat(r0,r1,r2) @ Wc) ---------
__global__ __launch_bounds__(128)
void gemm_classify_softmax(const float* __restrict__ r0, const float* __restrict__ r1,
                           const float* __restrict__ r2, const _Float16* __restrict__ Wf,
                           float* __restrict__ out) {
  __shared__ __align__(32) _Float16 sAf[5 * 32 * 16];
  __shared__ float logits[80 * 64];
  const int tid = threadIdx.x, wave = tid >> 5, lane = tid & 31;
  const int row0 = blockIdx.x * 80;
  v8f acc[5] = {};
  for (int kt = 0; kt < 14; ++kt) {            // K = 448
    // Segment select is uniform per k-step (tiles never straddle 64/192).
    const float* src; int ldshift, kb;
    if (kt < 2)      { src = r0; ldshift = 6; kb = kt * 32; }
    else if (kt < 6) { src = r1; ldshift = 7; kb = kt * 32 - 64; }
    else             { src = r2; ldshift = 8; kb = kt * 32 - 192; }
    load_Atile(sAf, src, row0, ldshift, kb, tid);
    v16h b = *(const v16h*)&Wf[(((unsigned)(kt * 4 + wave)) * 32 + lane) * 16];
    __syncthreads();
    wmma_5x(sAf, b, lane, acc);
    __syncthreads();
  }
  const int lo = lane >> 4, mr = lane & 15;
#pragma unroll
  for (int st = 0; st < 5; ++st)
#pragma unroll
    for (int e = 0; e < 8; ++e)
      logits[(st * 16 + e + (lo ? 8 : 0)) * 64 + (wave << 4) + mr] = acc[st][e];
  __syncthreads();
  if (tid < 80) {                     // one thread per row: stable softmax(64)
    const float* lrow = &logits[tid * 64];
    float mx = lrow[0];
    for (int c = 1; c < 64; ++c) mx = fmaxf(mx, lrow[c]);
    float sum = 0.f;
    for (int c = 0; c < 64; ++c) sum += __expf(lrow[c] - mx);
    float inv = 1.f / sum;
    float* orow = &out[(size_t)(row0 + tid) * 64];
    for (int c = 0; c < 64; ++c) orow[c] = __expf(lrow[c] - mx) * inv;
  }
}

// ------------------------------------------------------------------- launch
extern "C" void kernel_launch(void* const* d_in, const int* in_sizes, int n_in,
                              void* d_out, int out_size, void* d_ws, size_t ws_size,
                              hipStream_t stream) {
  const float* x       = (const float*)d_in[0];
  const int*   erow    = (const int*)  d_in[1];
  const int*   ecol    = (const int*)  d_in[2];
  const float* eval    = (const float*)d_in[3];
  const float* w_embed = (const float*)d_in[4];
  const float* w_cls   = (const float*)d_in[5];
  float* out = (float*)d_out;

  // workspace (floats): r0 | r1 | r2 | s | t | wf_e | wf_c   (~141 MB)
  float* ws = (float*)d_ws;
  float* r0 = ws;
  float* r1 = r0 + (size_t)GCN_N * 64;
  float* r2 = r1 + (size_t)GCN_N * 128;
  float* s  = r2 + (size_t)GCN_N * 256;
  float* t  = s  + (size_t)GCN_N * 128;
  _Float16* wf_e = (_Float16*)(t + (size_t)GCN_N * 128);  // 256*64 halfs
  _Float16* wf_c = wf_e + (size_t)GCN_FEAT * 64;          // 448*64 halfs

  // 0) pre-fragment both weight matrices (tiny one-off passes)
  prep_wfrag<<<(GCN_FEAT * 64 + 255) / 256, 256, 0, stream>>>(w_embed, wf_e, GCN_FEAT);
  prep_wfrag<<<(448 * 64 + 255) / 256, 256, 0, stream>>>(w_cls, wf_c, 448);

  // 1) r0 = relu(X @ We)
  gemm_embed_relu<<<GCN_N / 80, 128, 0, stream>>>(x, wf_e, r0);

  // 2) hop 1 (d=64): s = A r0, t = A s, r1 = relu([s-r0, t-s-r0])
  {
    int n4 = GCN_N * 64 / 4, e4 = GCN_E * 16;
    zero_f32v4<<<(n4 + 255) / 256, 256, 0, stream>>>((float4*)s, n4);
    zero_f32v4<<<(n4 + 255) / 256, 256, 0, stream>>>((float4*)t, n4);
    spmm_atomic<64><<<(e4 + 255) / 256, 256, 0, stream>>>(erow, ecol, eval, r0, s);
    spmm_atomic<64><<<(e4 + 255) / 256, 256, 0, stream>>>(erow, ecol, eval, s, t);
    hop_combine<64><<<(n4 + 255) / 256, 256, 0, stream>>>(r0, s, t, r1);
  }
  // 3) hop 2 (d=128): r2 = relu([s-r1, t-s-r1])
  {
    int n4 = GCN_N * 128 / 4, e4 = GCN_E * 32;
    zero_f32v4<<<(n4 + 255) / 256, 256, 0, stream>>>((float4*)s, n4);
    zero_f32v4<<<(n4 + 255) / 256, 256, 0, stream>>>((float4*)t, n4);
    spmm_atomic<128><<<(e4 + 255) / 256, 256, 0, stream>>>(erow, ecol, eval, r1, s);
    spmm_atomic<128><<<(e4 + 255) / 256, 256, 0, stream>>>(erow, ecol, eval, s, t);
    hop_combine<128><<<(n4 + 255) / 256, 256, 0, stream>>>(r1, s, t, r2);
  }

  // 4) out = softmax(concat(r0,r1,r2) @ Wc)
  gemm_classify_softmax<<<GCN_N / 80, 128, 0, stream>>>(r0, r1, r2, wf_c, out);
}
